// MyLSTMCell_29600914604232
// MI455X (gfx1250) — compile-verified, matrix operand order checked by reading, not run on hardware
//
#include <hip/hip_runtime.h>
#include <cstdint>

#define BATCH 4096
#define ISZ   1024
#define HSZ   2048
#define DSZ   (ISZ + HSZ)     // 3072
#define BM    128
#define BN    64              // per-gate N tile
#define BK    32
#define LDSTR 20              // padded LDS row stride in dwords (16 data + 4 pad)
#define NSTEPS (DSZ / BK)     // 96
#define HSTEPS (HSZ / BK)     // 64 (k-tiles sourced from h_prev)
#define LDSSZ  (BM * LDSTR + 4 * BN * LDSTR)   // 7680 dwords per buffer

typedef __attribute__((ext_vector_type(16))) __bf16       v16bf;
typedef __attribute__((ext_vector_type(8)))  float        v8f;
typedef __attribute__((ext_vector_type(8)))  unsigned int v8u;

__device__ __forceinline__ uint32_t pk_bf16(float lo, float hi) {
  unsigned short l = __builtin_bit_cast(unsigned short, (__bf16)lo);
  unsigned short h = __builtin_bit_cast(unsigned short, (__bf16)hi);
  return (uint32_t)l | ((uint32_t)h << 16);
}

__device__ __forceinline__ float sigmoid_(float x) {
  return 1.0f / (1.0f + __expf(-x));
}
__device__ __forceinline__ float tanh_(float x) {
  float e = __expf(2.0f * x);
  return (e - 1.0f) / (e + 1.0f);
}

__global__ __launch_bounds__(256)
void lstm_cell_wmma(const float* __restrict__ x,
                    const float* __restrict__ h_prev,
                    const float* __restrict__ c_prev,
                    const float* __restrict__ Wf, const float* __restrict__ bf,
                    const float* __restrict__ Wi, const float* __restrict__ bi,
                    const float* __restrict__ Wo, const float* __restrict__ bo,
                    const float* __restrict__ Wc, const float* __restrict__ bc,
                    float* __restrict__ out) {
  __shared__ __align__(16) uint32_t smem[2 * LDSSZ];   // double-buffered: 60 KB

  const int tid  = threadIdx.x;
  const int lane = tid & 31;
  const int wave = tid >> 5;
  const int half = lane >> 4;          // 0: lanes 0-15, 1: lanes 16-31
  const int lrow = lane & 15;

  const int mBase = blockIdx.x * BM;   // batch rows
  const int nBase = blockIdx.y * BN;   // per-gate hidden cols

  const float* Wg[4] = {Wf, Wi, Wo, Wc};

  const int wm = (wave & 3) * 32;      // wave tile: 32 rows
  const int wn = (wave >> 2) * 32;     // 32 cols per gate

  v8f acc[4][2][2];
#pragma unroll
  for (int g = 0; g < 4; ++g)
#pragma unroll
    for (int mi = 0; mi < 2; ++mi)
#pragma unroll
      for (int ni = 0; ni < 2; ++ni)
        acc[g][mi][ni] = (v8f){0.f, 0.f, 0.f, 0.f, 0.f, 0.f, 0.f, 0.f};

  // staging registers for the software pipeline
  uint32_t stA[8];
  uint32_t stB[4][4];

  const int arow = tid >> 1;           // 0..127
  const int aseg = tid & 1;            // 16-float segment within BK
  const int bn   = tid & 63;           // 0..63 column within B tile
  const int kd0  = (tid >> 6) * 4;     // packed-K dword base 0/4/8/12

  // ---- pointer accumulators: no per-iteration 64-bit address rebuilds ----
  const float* asrc = h_prev + (size_t)(mBase + arow) * HSZ + aseg * 16;
  const float* const xbase = x + (size_t)(mBase + arow) * ISZ + aseg * 16;
  const float* wsrc[4];
#pragma unroll
  for (int g = 0; g < 4; ++g)
    wsrc[g] = Wg[g] + (size_t)(2 * kd0) * HSZ + nBase + bn;

  auto stage = [&]() {   // global fp32 -> packed bf16 regs, for tile at current ptrs
#pragma unroll
    for (int j = 0; j < 4; ++j) {
      float4 v = ((const float4*)asrc)[j];
      stA[2 * j + 0] = pk_bf16(v.x, v.y);
      stA[2 * j + 1] = pk_bf16(v.z, v.w);
    }
#pragma unroll
    for (int g = 0; g < 4; ++g) {
      const float* w = wsrc[g];
#pragma unroll
      for (int j = 0; j < 4; ++j) {
        float w0 = w[(2 * j + 0) * HSZ];
        float w1 = w[(2 * j + 1) * HSZ];
        stB[g][j] = pk_bf16(w0, w1);
      }
    }
    // pull the following k-tile toward L2 (speculative global_prefetch_b8)
    __builtin_prefetch(asrc + BK, 0, 1);
#pragma unroll
    for (int g = 0; g < 4; ++g)
      __builtin_prefetch(wsrc[g] + (size_t)BK * HSZ, 0, 1);
  };

  auto advance = [&](int ktNextNext) {   // set pointers for tile ktNextNext
    if (ktNextNext == HSTEPS) asrc = xbase;   // cross the h_prev | x seam
    else                      asrc += BK;
#pragma unroll
    for (int g = 0; g < 4; ++g) wsrc[g] += (size_t)BK * HSZ;
  };

  auto commit = [&](uint32_t* base) {
    uint32_t* Al = base;
    uint32_t* Bl = base + BM * LDSTR;
    uint32_t* ad = Al + arow * LDSTR + aseg * 8;
    uint4 v0; v0.x = stA[0]; v0.y = stA[1]; v0.z = stA[2]; v0.w = stA[3];
    uint4 v1; v1.x = stA[4]; v1.y = stA[5]; v1.z = stA[6]; v1.w = stA[7];
    ((uint4*)ad)[0] = v0;
    ((uint4*)ad)[1] = v1;
#pragma unroll
    for (int g = 0; g < 4; ++g) {
      uint4 w; w.x = stB[g][0]; w.y = stB[g][1]; w.z = stB[g][2]; w.w = stB[g][3];
      *((uint4*)(Bl + (g * BN + bn) * LDSTR + kd0)) = w;
    }
  };

  // prologue: fill buffer 0 with tile 0
  stage();
  commit(smem);
  advance(1);
  __syncthreads();

  for (int kt = 0; kt < NSTEPS; ++kt) {
    uint32_t* cur = smem + (kt & 1) * LDSSZ;
    uint32_t* nxt = smem + ((kt + 1) & 1) * LDSSZ;
    const bool more = (kt + 1 < NSTEPS);

    if (more) stage();   // issue global loads for tile kt+1; no wait yet

    // ---- compute tile kt from cur ----
    const uint32_t* Al = cur;
    const uint32_t* Bl = cur + BM * LDSTR;

    // A fragments: 16-bit A 16x32 layout (lanes 0-15: K0-7,K16-23; 16-31: K8-15,K24-31)
    v16bf afr[2];
#pragma unroll
    for (int mi = 0; mi < 2; ++mi) {
      const uint32_t* ap = Al + (wm + mi * 16 + lrow) * LDSTR + half * 4;
      uint4 lo = *(const uint4*)ap;
      uint4 hi = *(const uint4*)(ap + 8);
      v8u t; t[0]=lo.x; t[1]=lo.y; t[2]=lo.z; t[3]=lo.w;
             t[4]=hi.x; t[5]=hi.y; t[6]=hi.z; t[7]=hi.w;
      afr[mi] = __builtin_bit_cast(v16bf, t);
    }
#pragma unroll
    for (int g = 0; g < 4; ++g) {
#pragma unroll
      for (int ni = 0; ni < 2; ++ni) {
        const uint32_t* bp = Bl + (g * BN + wn + ni * 16 + lrow) * LDSTR + half * 4;
        uint4 lo = *(const uint4*)bp;
        uint4 hi = *(const uint4*)(bp + 8);
        v8u t; t[0]=lo.x; t[1]=lo.y; t[2]=lo.z; t[3]=lo.w;
               t[4]=hi.x; t[5]=hi.y; t[6]=hi.z; t[7]=hi.w;
        v16bf bfr = __builtin_bit_cast(v16bf, t);
#pragma unroll
        for (int mi = 0; mi < 2; ++mi)
          acc[g][mi][ni] = __builtin_amdgcn_wmma_f32_16x16x32_bf16(
              false, afr[mi], false, bfr, (short)0, acc[g][mi][ni], false, false);
      }
    }

    // ---- commit tile kt+1 into the other buffer; single barrier per step ----
    if (more) {
      commit(nxt);
      advance(kt + 2);
    }
    __syncthreads();
  }

  // ---- fused LSTM epilogue; C/D layout: VGPR j -> M = base + j + half*8, N = lrow ----
  float* outH = out;
  float* outC = out + (size_t)BATCH * HSZ;
#pragma unroll
  for (int mi = 0; mi < 2; ++mi) {
#pragma unroll
    for (int ni = 0; ni < 2; ++ni) {
      const int n = nBase + wn + ni * 16 + lrow;
      const float vbf = bf[n], vbi = bi[n], vbo = bo[n], vbc = bc[n];
      const int m0 = mBase + wm + mi * 16 + half * 8;
#pragma unroll
      for (int j = 0; j < 8; ++j) {
        const size_t idx = (size_t)(m0 + j) * HSZ + n;
        const float cp = c_prev[idx];
        const float fg = sigmoid_(acc[0][mi][ni][j] + vbf);
        const float ig = sigmoid_(acc[1][mi][ni][j] + vbi);
        const float og = sigmoid_(acc[2][mi][ni][j] + vbo);
        const float cd = tanh_(acc[3][mi][ni][j] + vbc);
        const float cn = fg * cp + ig * cd;
        outH[idx] = og * tanh_(cn);
        outC[idx] = cn;
      }
    }
  }
}

extern "C" void kernel_launch(void* const* d_in, const int* in_sizes, int n_in,
                              void* d_out, int out_size, void* d_ws, size_t ws_size,
                              hipStream_t stream) {
  (void)in_sizes; (void)n_in; (void)out_size; (void)d_ws; (void)ws_size;
  const float* x      = (const float*)d_in[0];
  const float* h_prev = (const float*)d_in[1];
  const float* c_prev = (const float*)d_in[2];
  const float* W_f    = (const float*)d_in[3];
  const float* b_f    = (const float*)d_in[4];
  const float* W_i    = (const float*)d_in[5];
  const float* b_i    = (const float*)d_in[6];
  const float* W_o    = (const float*)d_in[7];
  const float* b_o    = (const float*)d_in[8];
  const float* W_c    = (const float*)d_in[9];
  const float* b_c    = (const float*)d_in[10];

  dim3 grid(BATCH / BM, HSZ / BN);   // (32, 32)
  lstm_cell_wmma<<<grid, 256, 0, stream>>>(x, h_prev, c_prev,
                                           W_f, b_f, W_i, b_i,
                                           W_o, b_o, W_c, b_c,
                                           (float*)d_out);
}
